// Corr_30099130811040
// MI455X (gfx1250) — matrix operand order repaired and checked by reading, not compiled, not run on hardware
//
#include <hip/hip_runtime.h>
#include <hip/hip_bf16.h>

typedef __attribute__((ext_vector_type(16))) __bf16 v16bf;
typedef __attribute__((ext_vector_type(8)))  float  v8f;

#define N_   2
#define C_   256
#define HW_  4096
#define K_   21
#define KP_  32
#define SCALE_ 0.21821789023599239f  /* 1/sqrt(21) */

union U8H { __bf16 h[8]; uint4 v; };

__device__ __forceinline__ v8f wmma_bf16(v16bf a, v16bf b, v8f c) {
  return __builtin_amdgcn_wmma_f32_16x16x32_bf16(false, a, false, b, (short)0, c,
                                                 false, false);
}

// A-fragment (16x32 bf16): per lane two contiguous 8-half runs (K=kb..kb+7, kb+16..kb+23)
__device__ __forceinline__ v16bf ld_a(const __bf16* lo, const __bf16* hi) {
  v16bf r;
#pragma unroll
  for (int i = 0; i < 8; ++i) { r[i] = lo[i]; r[8 + i] = hi[i]; }
  return r;
}
// B-fragment (32x16 bf16): per lane two 8-half runs (K=kb..kb+7, kb+16..kb+23 for fixed column)
__device__ __forceinline__ v16bf ld_b16(const __bf16* p) {
  v16bf r;
#pragma unroll
  for (int i = 0; i < 16; ++i) r[i] = p[i < 8 ? i : i + 8];
  return r;
}

// ---------------- prep: f1t/f2t = (w*feat+b) in bf16, K padded 21->32, k-contiguous ----
__global__ void k_prep_f(const float* __restrict__ feat,
                         const float* __restrict__ w1, const float* __restrict__ b1,
                         const float* __restrict__ w2, const float* __restrict__ b2,
                         __bf16* __restrict__ f1t, __bf16* __restrict__ f2t) {
  int gid = blockIdx.x * blockDim.x + threadIdx.x;   // N*HW threads
  int n = gid / HW_;
  int p = gid % HW_;
  float a1[K_], a2[K_];
#pragma unroll
  for (int k = 0; k < K_; ++k) { a1[k] = b1[k]; a2[k] = b2[k]; }
  const float* fp = feat + (size_t)n * C_ * HW_ + p;
  for (int c = 0; c < C_; ++c) {
    float fv = fp[(size_t)c * HW_];
#pragma unroll
    for (int k = 0; k < K_; ++k) {
      a1[k] = fmaf(w1[k * C_ + c], fv, a1[k]);
      a2[k] = fmaf(w2[k * C_ + c], fv, a2[k]);
    }
  }
  __bf16* o1 = f1t + (size_t)gid * KP_;
  __bf16* o2 = f2t + (size_t)gid * KP_;
#pragma unroll
  for (int k = 0; k < KP_; ++k) {
    o1[k] = (k < K_) ? (__bf16)(a1[k] * SCALE_) : (__bf16)0.0f;  // fold 1/sqrt(21)
    o2[k] = (k < K_) ? (__bf16)(a2[k])          : (__bf16)0.0f;
  }
}

// ---------------- prep: out_rb = bilinear(out, 128->64) bf16, classes padded to 32 -----
__global__ void k_prep_out(const float* __restrict__ outp, __bf16* __restrict__ out_rb) {
  int gid = blockIdx.x * blockDim.x + threadIdx.x;   // N*KP*HW threads
  int n = gid / (KP_ * HW_);
  int k = (gid / HW_) % KP_;
  int p = gid % HW_;
  float v = 0.0f;
  if (k < K_) {
    int y = p >> 6, x = p & 63;
    float fy = y * (127.0f / 63.0f);
    float fx = x * (127.0f / 63.0f);
    int y0 = (int)fy, x0 = (int)fx;
    int y1 = min(y0 + 1, 127), x1 = min(x0 + 1, 127);
    float wy = fy - (float)y0, wx = fx - (float)x0;
    const float* b = outp + ((size_t)(n * K_ + k)) * 128 * 128;
    float t = b[y0 * 128 + x0] * (1.0f - wy) + b[y1 * 128 + x0] * wy;
    float u = b[y0 * 128 + x1] * (1.0f - wy) + b[y1 * 128 + x1] * wy;
    v = t * (1.0f - wx) + u * wx;
  }
  out_rb[gid] = (__bf16)v;
}

// ---------------- pass 1: softmax row offsets off = rowmax + log(rowsum) ---------------
// Two passes over q (WMMA recompute is ~free): max-only pass, then 1-exp-per-element
// sum pass with independent exps (co-execute with XDL), instead of 2 exps + serial chain.
__global__ void __launch_bounds__(256)
k_stats(const __bf16* __restrict__ f1t, const __bf16* __restrict__ f2t,
        float* __restrict__ off) {
  __shared__ float partB[128];   // [16 rows][8 waves]
  __shared__ float rowM[16];
  int n  = blockIdx.x >> 8;
  int p0 = (blockIdx.x & 255) << 4;
  int t = threadIdx.x;
  int lane = t & 31, wv = t >> 5;
  int m = lane & 15, half = lane >> 4;
  int kbA = half * 8, kbB = half * 8;
  const __bf16* f1b = f1t + (size_t)n * HW_ * KP_;
  const __bf16* f2b = f2t + (size_t)n * HW_ * KP_;
  const __bf16* arow = f1b + (size_t)(p0 + m) * KP_;
  v16bf a = ld_a(arow + kbA, arow + kbA + 16);

  // ---- pass A: row max over the wave-private 512-wide q slice ----
  float pm[8];
#pragma unroll
  for (int r = 0; r < 8; ++r) pm[r] = -3.0e38f;
  for (int tq = 0; tq < 32; ++tq) {
    int q = wv * 512 + tq * 16 + m;
    v16bf b = ld_b16(f2b + (size_t)q * KP_ + kbB);
    v8f c = {};
    c = wmma_bf16(a, b, c);
#pragma unroll
    for (int r = 0; r < 8; ++r) pm[r] = fmaxf(pm[r], c[r]);
  }
#pragma unroll
  for (int r = 0; r < 8; ++r) {
#pragma unroll
    for (int d = 1; d < 16; d <<= 1)
      pm[r] = fmaxf(pm[r], __shfl_xor(pm[r], d, 32));
  }
  if (m == 0) {
#pragma unroll
    for (int r = 0; r < 8; ++r) partB[(r + half * 8) * 8 + wv] = pm[r];
  }
  __syncthreads();
  if (t < 16) {
    float M = -3.0e38f;
    for (int w = 0; w < 8; ++w) M = fmaxf(M, partB[t * 8 + w]);
    rowM[t] = M;
  }
  __syncthreads();
  float Mr[8];
#pragma unroll
  for (int r = 0; r < 8; ++r) Mr[r] = rowM[r + half * 8];

  // ---- pass B: sum of exp(logit - M), one exp per element, all independent ----
  float ps[8];
#pragma unroll
  for (int r = 0; r < 8; ++r) ps[r] = 0.0f;
  for (int tq = 0; tq < 32; ++tq) {
    int q = wv * 512 + tq * 16 + m;
    v16bf b = ld_b16(f2b + (size_t)q * KP_ + kbB);
    v8f c = {};
    c = wmma_bf16(a, b, c);
#pragma unroll
    for (int r = 0; r < 8; ++r) ps[r] += __expf(c[r] - Mr[r]);
  }
#pragma unroll
  for (int r = 0; r < 8; ++r) {
#pragma unroll
    for (int d = 1; d < 16; d <<= 1) ps[r] += __shfl_xor(ps[r], d, 32);
  }
  if (m == 0) {
#pragma unroll
    for (int r = 0; r < 8; ++r) partB[(r + half * 8) * 8 + wv] = ps[r];
  }
  __syncthreads();
  if (t < 16) {
    float S = 0.0f;
    for (int w = 0; w < 8; ++w) S += partB[t * 8 + w];
    off[n * HW_ + p0 + t] = rowM[t] + __logf(S);  // prob = exp(logit - off)
  }
}

// ---------------- pass 2: corr_out[n,c,q] = sum_p out_r[c,p]*softmax(p,q), all WMMA ----
__global__ void __launch_bounds__(256)
k_corrout(const __bf16* __restrict__ f1t, const __bf16* __restrict__ f2t,
          const __bf16* __restrict__ out_rb,
          const float* __restrict__ off,
          float* __restrict__ corr_out) {
  __shared__ float offS[HW_];                // 16KB
  // per-wave P-tile staged in B layout: [col][k], row padded to 40 halfs (80B = 20 banks,
  // conflict-free across the 16 columns); b128 LDS ops only.
  __shared__ uint4 stageRaw[8][16][5];       // 10KB
  __shared__ float acc[KP_ * 16];            // [32 classes][16 q] cross-wave reduce
  int n  = blockIdx.x >> 8;
  int q0 = (blockIdx.x & 255) << 4;
  int t = threadIdx.x;
  int lane = t & 31, wv = t >> 5;
  int m = lane & 15, half = lane >> 4;
  int kbA = half * 8, kbB = half * 8;
  for (int i = t; i < HW_; i += 256) offS[i] = off[n * HW_ + i];
  acc[t] = 0.0f; acc[t + 256] = 0.0f;
  __syncthreads();
  const __bf16* f1b = f1t + (size_t)n * HW_ * KP_;
  const __bf16* f2b = f2t + (size_t)n * HW_ * KP_;
  const __bf16* ob  = out_rb + (size_t)n * KP_ * HW_;
  v16bf bq = ld_b16(f2b + (size_t)(q0 + m) * KP_ + kbB);  // fixed q-tile B for logits
  v8f acc0 = {}, acc1 = {};
  for (int j = 0; j < 16; ++j) {             // wave-private 512-wide p slice, chunks of 32
    int pc = wv * 512 + j * 32;
    const __bf16* alo = f1b + (size_t)(pc + m) * KP_;
    const __bf16* ahi = f1b + (size_t)(pc + 16 + m) * KP_;
    v16bf a_lo = ld_a(alo + kbA, alo + kbA + 16);
    v16bf a_hi = ld_a(ahi + kbA, ahi + kbA + 16);
    v8f c0 = {}, c1 = {};
    c0 = wmma_bf16(a_lo, bq, c0);            // logits rows pc..pc+15
    c1 = wmma_bf16(a_hi, bq, c1);            // logits rows pc+16..pc+31
    // probabilities for this 32-row chunk, packed as two contiguous 8-half runs
    U8H u0, u1;
#pragma unroll
    for (int r = 0; r < 8; ++r) {
      int k0 = half * 8 + r;                 // c0 -> k = half*8+r
      int k1 = 16 + half * 8 + r;            // c1 -> k = 16+half*8+r
      u0.h[r] = (__bf16)__expf(c0[r] - offS[pc + k0]);
      u1.h[r] = (__bf16)__expf(c1[r] - offS[pc + k1]);
    }
    __syncthreads();                         // protect previous stage reads (WAR)
    stageRaw[wv][m][half]     = u0.v;        // k in [half*8,   half*8+8)
    stageRaw[wv][m][2 + half] = u1.v;        // k in [16+half*8, ...+8)
    __syncthreads();                         // publish probs for B reads
    U8H r0, r1;
    r0.v = stageRaw[wv][m][half];            // k in [half*8,   half*8+8)
    r1.v = stageRaw[wv][m][2 + half];        // k in [16+half*8, ...+8)
    v16bf b2;
#pragma unroll
    for (int i = 0; i < 8; ++i) { b2[i] = r0.h[i]; b2[8 + i] = r1.h[i]; }
    const __bf16* a0p = ob + (size_t)m * HW_ + pc;          // classes 0..15
    const __bf16* a1p = ob + (size_t)(m + 16) * HW_ + pc;   // classes 16..31 (padded 0)
    v16bf a2_0 = ld_a(a0p + kbA, a0p + kbA + 16);
    v16bf a2_1 = ld_a(a1p + kbA, a1p + kbA + 16);
    acc0 = wmma_bf16(a2_0, b2, acc0);
    acc1 = wmma_bf16(a2_1, b2, acc1);
  }
#pragma unroll
  for (int r = 0; r < 8; ++r) {
    int row = r + half * 8;
    atomicAdd(&acc[row * 16 + m],        acc0[r]);
    atomicAdd(&acc[(16 + row) * 16 + m], acc1[r]);
  }
  __syncthreads();
  if (t < K_ * 16) {
    int c = t >> 4, col = t & 15;
    corr_out[((size_t)n * K_ + c) * HW_ + q0 + col] = acc[c * 16 + col];
  }
}

// ---------------- sampled rows: recompute softmax row, resize 64x64->128x128, threshold -
__device__ __forceinline__ float bil128(const float* r, int idx) {
  int oy = idx >> 7, ox = idx & 127;
  float fy = oy * (63.0f / 127.0f);
  float fx = ox * (63.0f / 127.0f);
  int y0 = (int)fy, x0 = (int)fx;
  int y1 = min(y0 + 1, 63), x1 = min(x0 + 1, 63);
  float wy = fy - (float)y0, wx = fx - (float)x0;
  float a = r[y0 * 64 + x0], b = r[y0 * 64 + x1];
  float c = r[y1 * 64 + x0], d = r[y1 * 64 + x1];
  return (a * (1.0f - wy) + c * wy) * (1.0f - wx) + (b * (1.0f - wy) + d * wy) * wx;
}

__global__ void __launch_bounds__(256)
k_sample(const __bf16* __restrict__ f1t, const __bf16* __restrict__ f2t,
         const float* __restrict__ off,
         const int* __restrict__ index, float* __restrict__ cmap) {
  __shared__ float row[HW_];
  __shared__ float f1r[K_];
  __shared__ float redn[256], redx[256];
  int n = blockIdx.x >> 7;
  int j = blockIdx.x & 127;
  int t = threadIdx.x;
  int p = index[j];
  const __bf16* f1p = f1t + ((size_t)n * HW_ + p) * KP_;
  if (t < K_) f1r[t] = (float)f1p[t];
  __syncthreads();
  float OFF = off[n * HW_ + p];
  for (int q = t; q < HW_; q += 256) {
    const __bf16* f2p = f2t + ((size_t)n * HW_ + q) * KP_;
    float s = 0.0f;
#pragma unroll
    for (int k = 0; k < K_; ++k) s = fmaf(f1r[k], (float)f2p[k], s);
    row[q] = __expf(s - OFF);
  }
  __syncthreads();
  float mn = 3.0e38f, mx = -3.0e38f;
  for (int i = 0; i < 64; ++i) {
    float v = bil128(row, t + i * 256);
    mn = fminf(mn, v);
    mx = fmaxf(mx, v);
  }
  redn[t] = mn; redx[t] = mx;
  __syncthreads();
  for (int s = 128; s > 0; s >>= 1) {
    if (t < s) {
      redn[t] = fminf(redn[t], redn[t + s]);
      redx[t] = fmaxf(redx[t], redx[t + s]);
    }
    __syncthreads();
  }
  float thr = 0.5f * (redn[0] + redx[0]);
  float* outp = cmap + (size_t)(n * 128 + j) * 16384;
  for (int i = 0; i < 64; ++i) {
    int idx = t + i * 256;
    float v = bil128(row, idx);
    outp[idx] = (v > thr) ? 1.0f : 0.0f;
  }
}

extern "C" void kernel_launch(void* const* d_in, const int* in_sizes, int n_in,
                              void* d_out, int out_size, void* d_ws, size_t ws_size,
                              hipStream_t stream) {
  const float* feat = (const float*)d_in[0];
  const float* outp = (const float*)d_in[1];
  const float* w1   = (const float*)d_in[2];
  const float* b1   = (const float*)d_in[3];
  const float* w2   = (const float*)d_in[4];
  const float* b2   = (const float*)d_in[5];
  const int*   idx  = (const int*)d_in[6];

  char* ws = (char*)d_ws;
  __bf16* f1t    = (__bf16*)(ws + 0);        // N*HW*KP bf16 = 512KB  (pre-scaled 1/sqrt21)
  __bf16* f2t    = (__bf16*)(ws + 524288);   // 512KB
  __bf16* out_rb = (__bf16*)(ws + 1048576);  // N*KP*HW bf16 = 512KB
  float*  off    = (float*)(ws + 1572864);   // N*HW f32 = 32KB : rowmax + log(rowsum)

  float* cmap     = (float*)d_out;                                   // (N,128,128,128)
  float* corr_out = (float*)d_out + (size_t)N_ * 128 * 128 * 128;    // (N,21,64,64)

  k_prep_f  <<<N_ * HW_ / 256,       256, 0, stream>>>(feat, w1, b1, w2, b2, f1t, f2t);
  k_prep_out<<<N_ * KP_ * HW_ / 256, 256, 0, stream>>>(outp, out_rb);
  k_stats   <<<N_ * 256,             256, 0, stream>>>(f1t, f2t, off);
  k_corrout <<<N_ * 256,             256, 0, stream>>>(f1t, f2t, out_rb, off, corr_out);
  k_sample  <<<N_ * 128,             256, 0, stream>>>(f1t, f2t, off, idx, cmap);
}
